// SelfIIWA_77610059038869
// MI455X (gfx1250) — matrix operand order repaired and checked by reading, not compiled, not run on hardware
//
#include <hip/hip_runtime.h>

// ---------------------------------------------------------------------------
// CDNA5 (gfx1250) implementation of the Swin-style block.
// fp32 master state lives in d_out; all GEMMs run bf16 WMMA w/ f32 accum.
// Weight tiles are staged to LDS with the Tensor Data Mover (double-buffered).
// ---------------------------------------------------------------------------

typedef __bf16 bf16_t;
typedef __attribute__((ext_vector_type(16))) __bf16 v16bf;
typedef __attribute__((ext_vector_type(8)))  __bf16 v8bf;
typedef __attribute__((ext_vector_type(8)))  float  v8f;
typedef __attribute__((ext_vector_type(4)))  unsigned int v4u;
typedef __attribute__((ext_vector_type(8)))  int v8i;
typedef __attribute__((ext_vector_type(4)))  int v4i;

#define SCALEF 0.17677669529663687f   // 32^-0.5

__device__ __forceinline__ v16bf combine16(v8bf lo, v8bf hi) {
  v16bf r;
#pragma unroll
  for (int i = 0; i < 8; ++i) { r[i] = lo[i]; r[i + 8] = hi[i]; }
  return r;
}

__device__ __forceinline__ v8f wmma_bf16(v16bf a, v16bf b, v8f c) {
#if defined(__gfx1250__)
  return __builtin_amdgcn_wmma_f32_16x16x32_bf16(false, a, false, b,
                                                 (short)0, c, false, false);
#else
  return c;
#endif
}

// TDM: load a 64-row x 64-col bf16 tile (rows stride K elems) into LDS,
// packed as 64 rows x 128 bytes. 2-D descriptor per ISA ch.8.
// This toolchain exposes the 6-arg builtin:
//   (uint32x4 g0, int32x8 g1, int32x4 g2, int32x4 g3, int32x8 g4, i32 cpol)
__device__ __forceinline__ void tdm_load_tile(const bf16_t* gsrc,
                                              unsigned lds_off, int K)
{
#if defined(__gfx1250__)
  unsigned long long ga = (unsigned long long)(size_t)gsrc;
  v4u g0;
  g0[0] = 1u;                                            // count=1 (user D#)
  g0[1] = lds_off;                                       // LDS byte address
  g0[2] = (unsigned)(ga & 0xFFFFFFFFu);                  // global_addr[31:0]
  g0[3] = (unsigned)((ga >> 32) & 0x01FFFFFFu)           // global_addr[56:32]
          | (2u << 30);                                  // type = 2 ("image")
  v8i g1;
  g1[0] = (int)(1u << 16);                               // data_size=1 -> 2B
  g1[1] = (int)(((unsigned)K & 0xFFFFu) << 16);          // tensor_dim0 lo16
  g1[2] = (int)(((unsigned)K >> 16) | (64u << 16));      // dim0 hi16 | dim1 lo16
  g1[3] = (int)(64u << 16);                              // tile_dim0 = 64
  g1[4] = 64;                                            // tile_dim1 = 64
  g1[5] = K;                                             // dim0_stride lo32
  g1[6] = 0;
  g1[7] = 0;
  v4i z4 = {0, 0, 0, 0};
  v8i z8 = {0, 0, 0, 0, 0, 0, 0, 0};
  __builtin_amdgcn_tensor_load_to_lds(g0, g1, z4, z4, z8, 0);
#else
  (void)gsrc; (void)lds_off; (void)K;
#endif
}

__device__ __forceinline__ void tdm_wait(int pending)
{
#if defined(__gfx1250__)
  if (pending) __builtin_amdgcn_s_wait_tensorcnt(1);
  else         __builtin_amdgcn_s_wait_tensorcnt(0);
#else
  (void)pending;
#endif
}

// ---------------------------------------------------------------------------
// Generic GEMM: out[M,N] = A[M,K](bf16) @ W[N,K]^T(bf16) + bias
// optional exact GELU, optional fp32 residual add, fp32 and/or bf16 output.
// Block = 128 threads (4 waves); block tile 64x64; wave tile 16x64.
// W tiles (64x64 bf16, 8 KB) TDM-staged to LDS, double-buffered.
// Requires K % 64 == 0, M % 64 == 0, N % 64 == 0.
// ---------------------------------------------------------------------------
__global__ __launch_bounds__(128)
void gemm_bf16_wmma(const bf16_t* __restrict__ A, int M, int K,
                    const bf16_t* __restrict__ W, int N,
                    const float* __restrict__ bias,
                    float* __restrict__ outf,
                    bf16_t* __restrict__ outb,
                    const float* __restrict__ resid,
                    int gelu)
{
  __shared__ bf16_t wt[2][64 * 64];

  const int lane  = threadIdx.x & 31;
  const int wv    = threadIdx.x >> 5;
  const int bn    = blockIdx.x * 64;
  const int bm    = blockIdx.y * 64 + wv * 16;
  const int lrow  = lane & 15;
  const int khalf = lane >> 4;

  const bf16_t* arow  = A + (size_t)(bm + lrow) * K;
  const bf16_t* wbase = W + (size_t)bn * K;

  v8f acc[4];
#pragma unroll
  for (int j = 0; j < 4; ++j) acc[j] = {};

  const int nkc = K >> 6;               // 64-wide K chunks
  if (threadIdx.x < 32)
    tdm_load_tile(wbase, (unsigned)(size_t)&wt[0][0], K);

  for (int kc = 0; kc < nkc; ++kc) {
    const int buf = kc & 1;
    if (threadIdx.x < 32) {
      if (kc + 1 < nkc) {
        tdm_load_tile(wbase + (size_t)(kc + 1) * 64,
                      (unsigned)(size_t)&wt[buf ^ 1][0], K);
        tdm_wait(1);                    // oldest (current buf) is complete
      } else {
        tdm_wait(0);
      }
    }
    __syncthreads();                    // tile visible to all waves

#pragma unroll
    for (int kk = 0; kk < 64; kk += 32) {
      const bf16_t* pa = arow + kc * 64 + kk + khalf * 16;
      v16bf a = combine16(*(const v8bf*)pa, *(const v8bf*)(pa + 8));
      __builtin_prefetch(arow + kc * 64 + kk + 64, 0, 3);   // WGP-scope
#pragma unroll
      for (int j = 0; j < 4; ++j) {
        const bf16_t* pb = &wt[buf][(16 * j + lrow) * 64 + kk + khalf * 16];
        v16bf b = combine16(*(const v8bf*)pb, *(const v8bf*)(pb + 8));
        acc[j] = wmma_bf16(a, b, acc[j]);
      }
    }
    __syncthreads();                    // all reads of buf done before reuse
  }

#pragma unroll
  for (int j = 0; j < 4; ++j) {
    const int col = bn + 16 * j + lrow;
    const float bv = bias ? bias[col] : 0.f;
#pragma unroll
    for (int r = 0; r < 8; ++r) {
      const long row = (long)bm + khalf * 8 + r;
      float v = acc[j][r] + bv;
      if (gelu) v = 0.5f * v * (1.f + erff(v * 0.70710678118654752f));
      const size_t idx = (size_t)row * N + col;
      if (resid) v += resid[idx];
      if (outf) outf[idx] = v;
      if (outb) outb[idx] = (bf16_t)v;
    }
  }
}

// ---------------------------------------------------------------------------
// Window attention: one block per (window, head). N=64 tokens, HD=32.
// S = (q k^T)*scale + bias -> softmax -> O = P V, accumulated into x.
// ---------------------------------------------------------------------------
__global__ __launch_bounds__(128)
void win_attn(const bf16_t* __restrict__ qkv, const float* __restrict__ biasbuf,
              float* __restrict__ x)
{
  const int h   = blockIdx.x & 7;
  const int wid = blockIdx.x >> 3;
  const int b   = wid >> 8;
  const int wy  = (wid >> 4) & 15;
  const int wx  = wid & 15;
  const int tid = threadIdx.x;
  const int lane  = tid & 31;
  const int wv    = tid >> 5;
  const int lrow  = lane & 15;
  const int khalf = lane >> 4;

  __shared__ float  s[64][68];
  __shared__ bf16_t pm[64][72];
  __shared__ bf16_t vt[32][72];

  const int base_t = b * 16384 + wy * 8 * 128 + wx * 8;
#define TOKW(n) (base_t + ((n) >> 3) * 128 + ((n) & 7))

  // stage V^T : vt[d][token]
  for (int i = tid; i < 64 * 32; i += 128) {
    const int n = i >> 5, d = i & 31;
    vt[d][n] = qkv[(size_t)TOKW(n) * 768 + 512 + h * 32 + d];
  }

  // S = q k^T
  {
    const int n = wv * 16 + lrow;
    const bf16_t* pq = qkv + (size_t)TOKW(n) * 768 + h * 32 + khalf * 16;
    v16bf a = combine16(*(const v8bf*)pq, *(const v8bf*)(pq + 8));
#pragma unroll
    for (int j = 0; j < 4; ++j) {
      const int m = 16 * j + lrow;
      const bf16_t* pk = qkv + (size_t)TOKW(m) * 768 + 256 + h * 32 + khalf * 16;
      v16bf bfr = *(const v16bf*)pk;
      v8f acc = {};
      acc = wmma_bf16(a, bfr, acc);
      const int col = 16 * j + lrow;
#pragma unroll
      for (int r = 0; r < 8; ++r) {
        const int row = wv * 16 + khalf * 8 + r;
        s[row][col] = acc[r] * SCALEF + biasbuf[h * 4096 + row * 64 + col];
      }
    }
  }
  __syncthreads();

  if (tid < 64) {
    const int row = tid;
    float mx = -1e30f;
    for (int c = 0; c < 64; ++c) mx = fmaxf(mx, s[row][c]);
    float sum = 0.f;
    for (int c = 0; c < 64; ++c) { float e = __expf(s[row][c] - mx); s[row][c] = e; sum += e; }
    const float inv = 1.f / sum;
    for (int c = 0; c < 64; ++c) pm[row][c] = (bf16_t)(s[row][c] * inv);
  }
  __syncthreads();

  // O = P V  (accumulate into residual x)
#pragma unroll
  for (int j = 0; j < 2; ++j) {
    v8f acc = {};
#pragma unroll
    for (int kb = 0; kb < 64; kb += 32) {
      const int n = wv * 16 + lrow;
      const bf16_t* pp = &pm[n][kb + khalf * 16];
      v16bf a = combine16(*(const v8bf*)pp, *(const v8bf*)(pp + 8));
      const int d = 16 * j + lrow;
      const bf16_t* pv = &vt[d][kb + khalf * 16];
      v16bf bfr = combine16(*(const v8bf*)pv, *(const v8bf*)(pv + 8));
      acc = wmma_bf16(a, bfr, acc);
    }
    const int d = 16 * j + lrow;
    const int c = h * 32 + d;
#pragma unroll
    for (int r = 0; r < 8; ++r) {
      const int row = wv * 16 + khalf * 8 + r;
      const size_t idx = (size_t)TOKW(row) * 256 + c;
      x[idx] += acc[r];
    }
  }
#undef TOKW
}

// ---------------------------------------------------------------------------
// Global attention: grid = b(8) * head(8) * rowblock(8 of 32 rows). Ng=256.
// ---------------------------------------------------------------------------
__global__ __launch_bounds__(64)
void glob_attn(const bf16_t* __restrict__ qkvg, float* __restrict__ out)
{
  const int blk = blockIdx.x;
  const int rb = blk & 7;
  const int h  = (blk >> 3) & 7;
  const int b  = blk >> 6;
  const int tid = threadIdx.x;
  const int lane  = tid & 31;
  const int wv    = tid >> 5;
  const int lrow  = lane & 15;
  const int khalf = lane >> 4;

  __shared__ float  s[32][264];
  __shared__ bf16_t vt[32][264];

  for (int i = tid; i < 32 * 256; i += 64) {
    const int d = i >> 8, k = i & 255;
    vt[d][k] = qkvg[(size_t)(b * 256 + k) * 768 + 512 + h * 32 + d];
  }

  const int n0 = rb * 32;
  {
    const int n = n0 + wv * 16 + lrow;
    const bf16_t* pq = qkvg + (size_t)(b * 256 + n) * 768 + h * 32 + khalf * 16;
    v16bf a = combine16(*(const v8bf*)pq, *(const v8bf*)(pq + 8));
    for (int j = 0; j < 16; ++j) {
      const int m = 16 * j + lrow;
      const bf16_t* pk = qkvg + (size_t)(b * 256 + m) * 768 + 256 + h * 32 + khalf * 16;
      v16bf bfr = *(const v16bf*)pk;
      v8f acc = {};
      acc = wmma_bf16(a, bfr, acc);
      const int col = 16 * j + lrow;
#pragma unroll
      for (int r = 0; r < 8; ++r)
        s[wv * 16 + khalf * 8 + r][col] = acc[r] * SCALEF;
    }
  }
  __syncthreads();

  if (tid < 32) {
    const int row = tid;
    float mx = -1e30f;
    for (int c = 0; c < 256; ++c) mx = fmaxf(mx, s[row][c]);
    float sum = 0.f;
    for (int c = 0; c < 256; ++c) { float e = __expf(s[row][c] - mx); s[row][c] = e; sum += e; }
    const float inv = 1.f / sum;
    bf16_t* prow = (bf16_t*)&s[row][0];     // in-place pack: 2c < 4c, safe ascending
    for (int c = 0; c < 256; ++c) prow[c] = (bf16_t)(s[row][c] * inv);
  }
  __syncthreads();

#pragma unroll
  for (int j = 0; j < 2; ++j) {
    v8f acc = {};
    for (int kb = 0; kb < 256; kb += 32) {
      const bf16_t* pp = (const bf16_t*)&s[wv * 16 + lrow][0] + kb + khalf * 16;
      v16bf a = combine16(*(const v8bf*)pp, *(const v8bf*)(pp + 8));
      const int d = 16 * j + lrow;
      const bf16_t* pv = &vt[d][kb + khalf * 16];
      v16bf bfr = combine16(*(const v8bf*)pv, *(const v8bf*)(pv + 8));
      acc = wmma_bf16(a, bfr, acc);
    }
    const int d = 16 * j + lrow;
    const int c = h * 32 + d;
#pragma unroll
    for (int r = 0; r < 8; ++r) {
      const int n = n0 + wv * 16 + khalf * 8 + r;
      out[(size_t)(b * 256 + n) * 256 + c] = acc[r];
    }
  }
}

// ---------------------------------------------------------------------------
// LayerNorm over C=256, one wave per token (wave32, 8 elements/lane).
// ---------------------------------------------------------------------------
__global__ __launch_bounds__(256)
void ln_kernel(const float* __restrict__ x, const float* __restrict__ g,
               const float* __restrict__ be, bf16_t* __restrict__ outb,
               float* __restrict__ outf, int T)
{
  const int wv = threadIdx.x >> 5;
  const int lane = threadIdx.x & 31;
  const int t = blockIdx.x * 8 + wv;
  if (t >= T) return;
  const float* row = x + (size_t)t * 256;
  float v[8], sum = 0.f, sq = 0.f;
#pragma unroll
  for (int i = 0; i < 8; ++i) { v[i] = row[lane * 8 + i]; sum += v[i]; sq += v[i] * v[i]; }
#pragma unroll
  for (int o = 16; o > 0; o >>= 1) { sum += __shfl_xor(sum, o, 32); sq += __shfl_xor(sq, o, 32); }
  const float mu = sum * (1.f / 256.f);
  const float var = sq * (1.f / 256.f) - mu * mu;
  const float r = rsqrtf(var + 1e-5f);
#pragma unroll
  for (int i = 0; i < 8; ++i) {
    const int c = lane * 8 + i;
    const float y = (v[i] - mu) * r * g[c] + be[c];
    if (outb) outb[(size_t)t * 256 + c] = (bf16_t)y;
    if (outf) outf[(size_t)t * 256 + c] = y;
  }
}

// ---------------------------------------------------------------------------
// Window average-pool + sine positional encoding.
// ---------------------------------------------------------------------------
__device__ __forceinline__ float possine(int wy, int wx, int c) {
  const float twoPi = 6.2831853071795864f;
  const float denom = 16.f + 1e-5f;
  const float val = (c < 128) ? ((float)(wy + 1) / denom * twoPi)
                              : ((float)(wx + 1) / denom * twoPi);
  const int e = c & 127;
  const float expo = (float)(2 * (e >> 1)) / 128.f;
  const float p = val / powf(10000.f, expo);
  return (e & 1) ? cosf(p) : sinf(p);
}

__global__ __launch_bounds__(64)
void pool_pos(const float* __restrict__ x, float* __restrict__ xg,
              bf16_t* __restrict__ xgb)
{
  const int wid = blockIdx.x;              // b*256 + wy*16 + wx
  const int b = wid >> 8;
  const int wy = (wid >> 4) & 15;
  const int wx = wid & 15;
  const size_t t0 = (size_t)b * 16384 + (size_t)wy * 8 * 128 + wx * 8;
  for (int c = threadIdx.x; c < 256; c += 64) {
    float sum = 0.f;
    for (int p = 0; p < 64; ++p)
      sum += x[(t0 + (p >> 3) * 128 + (p & 7)) * 256 + c];
    const float v = sum * (1.f / 64.f) + possine(wy, wx, c);
    const size_t o = (size_t)wid * 256 + c;
    xg[o] = v; xgb[o] = (bf16_t)v;
  }
}

// ---------------------------------------------------------------------------
// Bilinear upsample (align_corners=True) 16x16 -> 128x128, accumulate into x.
// ---------------------------------------------------------------------------
__global__ __launch_bounds__(256)
void upsample_add(const float* __restrict__ xg2, float* __restrict__ x)
{
  const size_t gid = (size_t)blockIdx.x * blockDim.x + threadIdx.x;
  if (gid >= (size_t)8 * 128 * 128 * 256) return;
  const int c = gid & 255;
  const size_t pix = gid >> 8;
  const int xx = pix & 127;
  const size_t p2 = pix >> 7;
  const int yy = p2 & 127;
  const int b = (int)(p2 >> 7);
  const float sy = yy * (15.f / 127.f);
  const int y0 = (int)sy; const float ty = sy - y0; const int y1 = min(y0 + 1, 15);
  const float sx = xx * (15.f / 127.f);
  const int x0 = (int)sx; const float tx = sx - x0; const int x1 = min(x0 + 1, 15);
  const float* g = xg2 + (size_t)b * 256 * 256;
#define GV(Y,X) g[(((Y) * 16 + (X)) * 256) + c]
  const float v = (1.f - ty) * ((1.f - tx) * GV(y0, x0) + tx * GV(y0, x1)) +
                  ty        * ((1.f - tx) * GV(y1, x0) + tx * GV(y1, x1));
#undef GV
  x[gid] += v;
}

// ---------------------------------------------------------------------------
// Small utility kernels.
// ---------------------------------------------------------------------------
__global__ __launch_bounds__(256)
void cast_f32_bf16(const float* __restrict__ in, bf16_t* __restrict__ out, long n)
{
  const long i = (long)blockIdx.x * blockDim.x + threadIdx.x;
  if (i < n) out[i] = (bf16_t)in[i];
}

__global__ __launch_bounds__(256)
void bias_expand(const float* __restrict__ rel_table, const int* __restrict__ rel_idx,
                 float* __restrict__ biasbuf)
{
  const int i = blockIdx.x * blockDim.x + threadIdx.x;
  if (i < 8 * 4096) {
    const int h = i >> 12;
    const int nm = i & 4095;
    biasbuf[i] = rel_table[rel_idx[nm] * 8 + h];
  }
}

// ---------------------------------------------------------------------------
// Host driver.
// ---------------------------------------------------------------------------
extern "C" void kernel_launch(void* const* d_in, const int* in_sizes, int n_in,
                              void* d_out, int out_size, void* d_ws, size_t ws_size,
                              hipStream_t stream)
{
  (void)in_sizes; (void)n_in; (void)out_size; (void)ws_size;

  const float* x_in       = (const float*)d_in[0];
  const float* rel_table  = (const float*)d_in[1];
  const float* qkv_w      = (const float*)d_in[2];
  const float* qkv_b      = (const float*)d_in[3];
  const float* qkv2_w     = (const float*)d_in[4];
  const float* qkv2_b     = (const float*)d_in[5];
  const float* proj_ln_w  = (const float*)d_in[6];
  const float* proj_ln_b  = (const float*)d_in[7];
  const float* proj_w     = (const float*)d_in[8];
  const float* proj_b     = (const float*)d_in[9];
  const float* proj2_ln_w = (const float*)d_in[10];
  const float* proj2_ln_b = (const float*)d_in[11];
  const float* proj2_w    = (const float*)d_in[12];
  const float* proj2_b    = (const float*)d_in[13];
  const float* norm1_w    = (const float*)d_in[14];
  const float* norm1_b    = (const float*)d_in[15];
  const float* norm2_w    = (const float*)d_in[16];
  const float* norm2_b    = (const float*)d_in[17];
  const float* fc1_w      = (const float*)d_in[18];
  const float* fc1_b      = (const float*)d_in[19];
  const float* fc2_w      = (const float*)d_in[20];
  const float* fc2_b      = (const float*)d_in[21];
  const int*   rel_idx    = (const int*)d_in[22];

  const long T = 8L * 128 * 128;          // 131072 tokens
  float* xst = (float*)d_out;             // fp32 master state

  // ---- workspace layout ----
  char* wsb = (char*)d_ws;
  size_t off = 0;
  auto take = [&](size_t bytes) -> char* {
    char* p = wsb + off;
    off = (off + bytes + 255) & ~(size_t)255;
    return p;
  };
  bf16_t* qkvw_b  = (bf16_t*)take(768 * 256 * 2);
  bf16_t* qkv2w_b = (bf16_t*)take(768 * 256 * 2);
  bf16_t* projw_b = (bf16_t*)take(256 * 256 * 2);
  bf16_t* proj2w_b= (bf16_t*)take(256 * 256 * 2);
  bf16_t* fc1w_b  = (bf16_t*)take(1024 * 256 * 2);
  bf16_t* fc2w_b  = (bf16_t*)take(256 * 1024 * 2);
  float*  biasbuf = (float*)take(8 * 4096 * 4);
  bf16_t* xbf     = (bf16_t*)take((size_t)T * 256 * 2);   //  67 MB
  bf16_t* big     = (bf16_t*)take((size_t)T * 1024 * 2);  // 268 MB (qkv / fc1 act)
  float*  xg      = (float*)take(2048 * 256 * 4);
  bf16_t* xgb     = (bf16_t*)take(2048 * 256 * 2);
  bf16_t* qkvg    = (bf16_t*)take(2048 * 768 * 2);
  float*  xg2     = (float*)take(2048 * 256 * 4);

  // ---- x state = input ----
  (void)hipMemcpyAsync(xst, x_in, (size_t)T * 256 * sizeof(float),
                       hipMemcpyDeviceToDevice, stream);

  // ---- weight casts + bias table ----
  auto cast = [&](const float* src, bf16_t* dst, long n) {
    cast_f32_bf16<<<(unsigned)((n + 255) / 256), 256, 0, stream>>>(src, dst, n);
  };
  cast(qkv_w,  qkvw_b,  768 * 256);
  cast(qkv2_w, qkv2w_b, 768 * 256);
  cast(proj_w, projw_b, 256 * 256);
  cast(proj2_w,proj2w_b,256 * 256);
  cast(fc1_w,  fc1w_b,  1024 * 256);
  cast(fc2_w,  fc2w_b,  256 * 1024);
  bias_expand<<<128, 256, 0, stream>>>(rel_table, rel_idx, biasbuf);

  // ---- 1) window attention branch ----
  cast(x_in, xbf, T * 256);
  gemm_bf16_wmma<<<dim3(768 / 64, (unsigned)(T / 64)), 128, 0, stream>>>(
      xbf, (int)T, 256, qkvw_b, 768, qkv_b, nullptr, big, nullptr, 0);
  win_attn<<<2048 * 8, 128, 0, stream>>>(big, biasbuf, xst);

  // ---- 2) proj (pre-LN, residual) ----
  ln_kernel<<<(unsigned)(T / 8), 256, 0, stream>>>(xst, proj_ln_w, proj_ln_b,
                                                   xbf, nullptr, (int)T);
  gemm_bf16_wmma<<<dim3(256 / 64, (unsigned)(T / 64)), 128, 0, stream>>>(
      xbf, (int)T, 256, projw_b, 256, proj_b, xst, nullptr, xst, 0);

  // ---- 3) global branch: pool + pos, qkv2, attention, upsample ----
  pool_pos<<<2048, 64, 0, stream>>>(xst, xg, xgb);
  gemm_bf16_wmma<<<dim3(768 / 64, 2048 / 64), 128, 0, stream>>>(
      xgb, 2048, 256, qkv2w_b, 768, qkv2_b, nullptr, qkvg, nullptr, 0);
  glob_attn<<<512, 64, 0, stream>>>(qkvg, xg2);
  upsample_add<<<(unsigned)((T * 256 + 255) / 256), 256, 0, stream>>>(xg2, xst);

  // ---- 4) proj2 (pre-LN, residual) ----
  ln_kernel<<<(unsigned)(T / 8), 256, 0, stream>>>(xst, proj2_ln_w, proj2_ln_b,
                                                   xbf, nullptr, (int)T);
  gemm_bf16_wmma<<<dim3(256 / 64, (unsigned)(T / 64)), 128, 0, stream>>>(
      xbf, (int)T, 256, proj2w_b, 256, proj2_b, xst, nullptr, xst, 0);

  // ---- 5) MLP: LN -> fc1 + GELU -> fc2 (residual) ----
  ln_kernel<<<(unsigned)(T / 8), 256, 0, stream>>>(xst, norm1_w, norm1_b,
                                                   xbf, nullptr, (int)T);
  gemm_bf16_wmma<<<dim3(1024 / 64, (unsigned)(T / 64)), 128, 0, stream>>>(
      xbf, (int)T, 256, fc1w_b, 1024, fc1_b, nullptr, big, nullptr, 1);
  gemm_bf16_wmma<<<dim3(256 / 64, (unsigned)(T / 64)), 128, 0, stream>>>(
      big, (int)T, 1024, fc2w_b, 256, fc2_b, xst, nullptr, xst, 0);

  // ---- 6) final LayerNorm in place on d_out ----
  ln_kernel<<<(unsigned)(T / 8), 256, 0, stream>>>(xst, norm2_w, norm2_b,
                                                   nullptr, xst, (int)T);
}
